// Stickbreaking_38646115729879
// MI455X (gfx1250) — matrix operand order repaired and checked by reading, not compiled, not run on hardware
//
#include <hip/hip_runtime.h>

typedef __attribute__((ext_vector_type(16))) __bf16 v16bf;
typedef __attribute__((ext_vector_type(2)))  __bf16 bf16x2;
typedef __attribute__((ext_vector_type(8)))  float  v8f;

#define SB_S  2048
#define SB_D  64
#define SB_H  16
#define SB_BM 64
#define SB_NW 4
#define NKB   (SB_S / 32)
#define QS_STR 72   // bf16 units, 64 + 8 pad
#define KF_STR 68   // f32 units, 64 + 4 pad
#define VT_STR 36   // f32 units, 32 + 4 pad
#define WS_STR 36   // bf16 units, 32 + 4 pad

#define L2E   1.4426950408889634f     // log2(e)
#define ZMSK  (-1.0e9f)               // mask added to z (linear domain)
#define LMSK2 (-1.4426950e9f)         // mask added to log2(beta) (= -1e9 * log2e)

__device__ __forceinline__ unsigned short bfbits(float f) {
    __bf16 b = (__bf16)f;                          // native v_cvt bf16 (RNE)
    return __builtin_bit_cast(unsigned short, b);
}
__device__ __forceinline__ unsigned packbf(float lo, float hi) {
    bf16x2 t; t.x = (__bf16)lo; t.y = (__bf16)hi;  // v_cvt_pk_bf16_f32 (confirmed)
    return __builtin_bit_cast(unsigned, t);
}

union ABReg { unsigned int u[8]; v16bf v; };

// One 32-key block of K (row-major) + V (transposed) f32 tiles via
// ASYNCcnt-tracked DMA into LDS. 4x B128 + 16x B32 = 20 issues per wave.
__device__ __forceinline__ void issue_kv_async(
    const float* __restrict__ kp, const float* __restrict__ vp,
    unsigned kf_lds, unsigned vt_lds, int wave, int lane)
{
    #pragma unroll
    for (int i = 0; i < 4; ++i) {                     // K: 512 float4s total
        int n16 = (wave * 4 + i) * 32 + lane;
        int row = n16 >> 4;
        int c4  = (n16 & 15) << 2;
        unsigned lds = kf_lds + (unsigned)((row * KF_STR + c4) * 4);
        unsigned long long ga =
            (unsigned long long)(uintptr_t)(kp + (size_t)row * SB_D + c4);
        asm volatile("global_load_async_to_lds_b128 %0, %1, off"
                     :: "v"(lds), "v"(ga) : "memory");
    }
    #pragma unroll
    for (int i = 0; i < 16; ++i) {                    // V: 2048 f32, scatter-transpose
        int n   = (wave * 16 + i) * 32 + lane;
        int row = n >> 6;                             // key j
        int d   = n & 63;
        unsigned lds = vt_lds + (unsigned)((d * VT_STR + row) * 4);
        unsigned long long ga =
            (unsigned long long)(uintptr_t)(vp + (size_t)row * SB_D + d);
        asm volatile("global_load_async_to_lds_b32 %0, %1, off"
                     :: "v"(lds), "v"(ga) : "memory");
    }
}

__global__ __launch_bounds__(128) void stickbreak_fused(
    const float* __restrict__ qg, const float* __restrict__ kg,
    const float* __restrict__ vg, float* __restrict__ outg)
{
    __shared__ float          Kf[2][32 * KF_STR];     // double-buffered K, f32
    __shared__ float          Vt[2][SB_D * VT_STR];   // double-buffered V^T, f32
    __shared__ unsigned short Qs[SB_BM * QS_STR];     // Q, bf16, pre-scaled
    __shared__ unsigned short Ws[SB_NW][16 * WS_STR]; // per-wave W scratch, bf16

    const int tid  = threadIdx.x;
    const int lane = tid & 31;
    const int wave = tid >> 5;
    const int m    = lane & 15;   // N / column index in WMMA layouts
    const int g    = lane >> 4;   // lane half

    const int h     = blockIdx.x / (SB_S / SB_BM);
    const int qtile = blockIdx.x % (SB_S / SB_BM);
    const int qbase = qtile * SB_BM;

    const float* kh = kg + (size_t)h * SB_S * SB_D;
    const float* vh = vg + (size_t)h * SB_S * SB_D;

    // ---- kick off DMA for the first (highest-j) key block into buffer 0 ----
    issue_kv_async(kh + (size_t)(NKB - 1) * 32 * SB_D,
                   vh + (size_t)(NKB - 1) * 32 * SB_D,
                   (unsigned)(uintptr_t)&Kf[0][0],
                   (unsigned)(uintptr_t)&Vt[0][0], wave, lane);

    // ---- stage Q (scaled by 1/sqrt(Dh)=0.125) as bf16 ----
    {
        const float* qh = qg + ((size_t)h * SB_S + qbase) * SB_D;
        #pragma unroll
        for (int it = 0; it < 8; ++it) {
            int idx = tid + it * 128;              // float4 index 0..1023
            int row = idx >> 4;
            int c4  = (idx & 15) << 2;
            float4 f = ((const float4*)(qh + (size_t)row * SB_D))[idx & 15];
            unsigned short* dst = &Qs[row * QS_STR + c4];
            dst[0] = bfbits(f.x * 0.125f);
            dst[1] = bfbits(f.y * 0.125f);
            dst[2] = bfbits(f.z * 0.125f);
            dst[3] = bfbits(f.w * 0.125f);
        }
    }
    __syncthreads();

    // ---- persistent per-wave Q A-operands: 16x64 -> two 16x32 bf16 ----
    ABReg aQ0, aQ1;
    {
        const unsigned int* prow = (const unsigned int*)&Qs[(wave * 16 + m) * QS_STR];
        #pragma unroll
        for (int d8 = 0; d8 < 8; ++d8) {
            int kk = 2 * d8 + 16 * g;  // A K contiguous (k = e + 16g)
            aQ0.u[d8] = prow[kk >> 1];
            aQ1.u[d8] = prow[(kk + 32) >> 1];
        }
    }

    v8f accO[4] = {};
    float carry[8];                      // running suffix sum, log2 domain
    #pragma unroll
    for (int r = 0; r < 8; ++r) carry[r] = 0.0f;

    // suffix order: iterate key blocks from high j to low j
    for (int kb = NKB - 1; kb >= 0; --kb) {
        const int ib = (NKB - 1 - kb) & 1;            // current buffer
        asm volatile("s_wait_asynccnt 0x0" ::: "memory");  // own DMA for cur buf done
        __syncthreads();   // all waves' DMA visible; all done reading other buf
        if (kb > 0) {      // prefetch next block (overlaps with compute below)
            issue_kv_async(kh + (size_t)(kb - 1) * 32 * SB_D,
                           vh + (size_t)(kb - 1) * 32 * SB_D,
                           (unsigned)(uintptr_t)&Kf[ib ^ 1][0],
                           (unsigned)(uintptr_t)&Vt[ib ^ 1][0], wave, lane);
        }

        const float* kbuf = &Kf[ib][0];
        const float* vbuf = &Vt[ib][0];

        // two 16-key subtiles, high j first (suffix-scan order)
        #pragma unroll
        for (int sub = 1; sub >= 0; --sub) {
            ABReg bK0, bK1;   // B 32x16: K = e + 16g (kdim), N = lane%16 (key)
            const float* prow = kbuf + (sub * 16 + m) * KF_STR;
            #pragma unroll
            for (int d8 = 0; d8 < 8; ++d8) {
                int col = 2 * d8 + 16 * g;
                float2 lo = *(const float2*)(prow + col);
                float2 hi = *(const float2*)(prow + col + 32);
                bK0.u[d8] = packbf(lo.x, lo.y);
                bK1.u[d8] = packbf(hi.x, hi.y);
            }
            v8f c = {};
            c = __builtin_amdgcn_wmma_f32_16x16x32_bf16(false, aQ0.v, false, bK0.v,
                                                        (short)0, c, false, false);
            c = __builtin_amdgcn_wmma_f32_16x16x32_bf16(false, aQ1.v, false, bK1.v,
                                                        (short)0, c, false, false);
            const int jg = kb * 32 + sub * 16 + m;
            #pragma unroll
            for (int r = 0; r < 8; ++r) {
                float s  = c[r];
                int  ig  = qbase + wave * 16 + r + 8 * g;
                // log2-domain stickbreaking: raw HW transcendentals, no fixups
                float en  = __builtin_amdgcn_exp2f(-L2E * s);      // exp(-s)
                float den = 1.0f + en;
                float sig = __builtin_amdgcn_rcpf(den);            // sigmoid(s)
                float sp2 = fmaf(s, L2E,
                                 __builtin_amdgcn_logf(den));      // softplus(s)*log2e
                bool  msk = (jg > ig);
                float z   = sig + (msk ? ZMSK : 0.0f);
                float sfx = (msk ? LMSK2 : 0.0f) - sp2;            // log2(beta)
                // inclusive suffix-sum over N within 16-lane halves
                #pragma unroll
                for (int dlt = 1; dlt < 16; dlt <<= 1) {
                    float o = __shfl_down(sfx, dlt, 16);
                    sfx += (m + dlt < 16) ? o : 0.0f;
                }
                float w = z * __builtin_amdgcn_exp2f(carry[r] + sfx);
                carry[r] += __shfl(sfx, g << 4, 32);               // tile total (N=0)
                Ws[wave][(r + 8 * g) * WS_STR + sub * 16 + m] = bfbits(w);
            }
        }

        // W (16x32) as A-operand, V columns as B-operands; accumulate O
        ABReg aW;
        {
            const unsigned int* prow = (const unsigned int*)&Ws[wave][m * WS_STR];
            #pragma unroll
            for (int d8 = 0; d8 < 8; ++d8) {
                int kk = 2 * d8 + 16 * g;
                aW.u[d8] = prow[kk >> 1];
            }
        }
        #pragma unroll
        for (int t = 0; t < 4; ++t) {
            ABReg bV;   // B 32x16: K = j local (= e+16g), N = d = 16t + lane%16
            const float* prow = vbuf + (t * 16 + m) * VT_STR;
            #pragma unroll
            for (int d8 = 0; d8 < 8; ++d8) {
                int col = 2 * d8 + 16 * g;
                float2 f = *(const float2*)(prow + col);
                bV.u[d8] = packbf(f.x, f.y);
            }
            accO[t] = __builtin_amdgcn_wmma_f32_16x16x32_bf16(false, aW.v, false, bV.v,
                                                              (short)0, accO[t], false, false);
        }
    }

    // ---- write out: C layout (row = r + 8g, col = 16t + m) ----
    float* oh = outg + ((size_t)h * SB_S + qbase + wave * 16) * SB_D;
    #pragma unroll
    for (int t = 0; t < 4; ++t) {
        #pragma unroll
        for (int r = 0; r < 8; ++r) {
            oh[(size_t)(r + 8 * g) * SB_D + t * 16 + m] = accO[t][r];
        }
    }
}

extern "C" void kernel_launch(void* const* d_in, const int* in_sizes, int n_in,
                              void* d_out, int out_size, void* d_ws, size_t ws_size,
                              hipStream_t stream) {
    (void)in_sizes; (void)n_in; (void)out_size; (void)d_ws; (void)ws_size;
    const float* q = (const float*)d_in[0];
    const float* k = (const float*)d_in[1];
    const float* v = (const float*)d_in[2];
    float* out = (float*)d_out;
    dim3 grid(SB_H * (SB_S / SB_BM));   // 16 heads x 32 query tiles = 512 WGs
    dim3 block(128);                    // 4 wave32, one 16-row tile per wave
    hipLaunchKernelGGL(stickbreak_fused, grid, block, 0, stream, q, k, v, out);
}
